// MultiHeadTokenAttention_7773890806120
// MI455X (gfx1250) — matrix operand order repaired and checked by reading, not compiled, not run on hardware
//
#include <hip/hip_runtime.h>

// ---------------------------------------------------------------------------
// MI455X (gfx1250) fused multi-head token attention.
//   D=1024, H=16, HD=64, N=64 tokens, T*B=1024 independent attention problems.
//   Pipeline:
//     1) transpose_convert : Wq/Wk/Wv f32[k][n] -> bf16 WT[n][k] (WMMA-B friendly)
//     2) qproj_kernel      : Q = query @ Wq + bq   (bf16 WMMA, f32 accum) -> ws
//     3) attn_kernel       : per (t,b): tokens tile -> bf16 LDS once; per head
//                            K_h/V_h via WMMA, softmax+attend in VALU.
//   attn_kernel wave assignment: each wave owns ONE column tile (unique B
//   fragment per k-step -> no duplicated weight traffic) and sweeps the 4 row
//   tiles with A fragments from LDS (cheap).
// ---------------------------------------------------------------------------

typedef __attribute__((ext_vector_type(16))) __bf16 v16bf;
typedef __attribute__((ext_vector_type(8)))  __bf16 v8bf;
typedef __attribute__((ext_vector_type(8)))  float  v8f;
typedef __attribute__((ext_vector_type(4)))  float  v4f;

#define DDIM   1024
#define NHEADS 16
#define HD     64
#define NTOK   64
#define XPAD   1032   // 64-row bf16 token tile row stride (bank-conflict free)
#define KVPAD  65     // K/V f32 tile row stride

__device__ __forceinline__ v16bf bf_combine(v8bf lo, v8bf hi) {
  v16bf r;
#pragma unroll
  for (int i = 0; i < 8; ++i) { r[i] = lo[i]; r[i + 8] = hi[i]; }
  return r;
}

// two 8-bf16 runs (p[0..7], p[16..23]) -> B fragment (table k order)
__device__ __forceinline__ v16bf load_b_frag(const __bf16* p) {
  return bf_combine(*(const v8bf*)p, *(const v8bf*)(p + 16));
}

// ---------------------------------------------------------------------------
// 1) f32 [k][n] -> bf16 [n][k] transpose+convert, 32x32 LDS tiles, 3 matrices
// ---------------------------------------------------------------------------
__global__ void transpose_convert(const float* __restrict__ s0,
                                  const float* __restrict__ s1,
                                  const float* __restrict__ s2,
                                  __bf16* __restrict__ d0,
                                  __bf16* __restrict__ d1,
                                  __bf16* __restrict__ d2) {
  const float* src = (blockIdx.z == 0) ? s0 : (blockIdx.z == 1) ? s1 : s2;
  __bf16*      dst = (blockIdx.z == 0) ? d0 : (blockIdx.z == 1) ? d1 : d2;
  __shared__ float tile[32][33];
  const int tx = threadIdx.x, ty = threadIdx.y;
#pragma unroll
  for (int i = ty; i < 32; i += 8) {
    tile[i][tx] = src[(blockIdx.y * 32 + i) * DDIM + blockIdx.x * 32 + tx];
  }
  __syncthreads();
#pragma unroll
  for (int i = ty; i < 32; i += 8) {
    dst[(blockIdx.x * 32 + i) * DDIM + blockIdx.y * 32 + tx] = (__bf16)tile[tx][i];
  }
}

// ---------------------------------------------------------------------------
// 2) Q projection: [1024,1024] @ WqT(bf16, n-major) -> f32 Qout. 128 thr = 4
//    waves; wave w owns row-tile w, sweeps 4 col-tiles; K loop 32 x (16x16x32).
// ---------------------------------------------------------------------------
__global__ void qproj_kernel(const float* __restrict__ Q,
                             const __bf16* __restrict__ WT,
                             const float* __restrict__ bias,
                             float* __restrict__ Qout) {
  const int lane  = threadIdx.x & 31;
  const int wv    = threadIdx.x >> 5;          // 0..3
  const int l15   = lane & 15;
  const int lhalf = (lane >> 4) & 1;
  const int rowBase = blockIdx.y * 64 + wv * 16;
  const int colBase = blockIdx.x * 64;

  v8f acc[4];
#pragma unroll
  for (int i = 0; i < 4; ++i)
#pragma unroll
    for (int j = 0; j < 8; ++j) acc[i][j] = 0.0f;

  const float* arow = Q + (size_t)(rowBase + l15) * DDIM;
  for (int kb = 0; kb < DDIM; kb += 32) {
    const int kOff = kb + lhalf * 8;
    v4f a0 = *(const v4f*)(arow + kOff);
    v4f a1 = *(const v4f*)(arow + kOff + 4);
    v4f a2 = *(const v4f*)(arow + kOff + 16);
    v4f a3 = *(const v4f*)(arow + kOff + 20);
    v16bf afrag;
#pragma unroll
    for (int i = 0; i < 4; ++i) {
      afrag[i]      = (__bf16)a0[i];
      afrag[4 + i]  = (__bf16)a1[i];
      afrag[8 + i]  = (__bf16)a2[i];
      afrag[12 + i] = (__bf16)a3[i];
    }
    const int kpos = kb + ((lane >= 16) ? 8 : 0);
#pragma unroll
    for (int nt = 0; nt < 4; ++nt) {
      v16bf bfrag = load_b_frag(WT + (size_t)(colBase + nt * 16 + l15) * DDIM + kpos);
      acc[nt] = __builtin_amdgcn_wmma_f32_16x16x32_bf16(
          false, afrag, false, bfrag, (short)0, acc[nt], false, false);
    }
  }
  const int rmo = (lane >= 16) ? 8 : 0;
#pragma unroll
  for (int nt = 0; nt < 4; ++nt) {
    const int col = colBase + nt * 16 + l15;
    const float b = bias[col];
#pragma unroll
    for (int r = 0; r < 8; ++r)
      Qout[(size_t)(rowBase + r + rmo) * DDIM + col] = acc[nt][r] + b;
  }
}

// ---------------------------------------------------------------------------
// 3) Fused per-(t,b) attention. 256 thr = 8 waves.
//    Waves 0-3 -> K matrix, waves 4-7 -> V matrix.
//    Wave owns column tile (wv&3): ONE B fragment per k-step (no duplicate
//    weight loads within the WG), sweeps row tiles 0..3 via LDS A fragments.
// ---------------------------------------------------------------------------
__global__ void attn_kernel(const float* __restrict__ tokens,
                            const float* __restrict__ gate,
                            const __bf16* __restrict__ WkT,
                            const __bf16* __restrict__ WvT,
                            const float* __restrict__ bk,
                            const float* __restrict__ bv,
                            const float* __restrict__ Qout,
                            float* __restrict__ attended,
                            float* __restrict__ mean_out,
                            float* __restrict__ logits_out) {
  __shared__ __bf16 Xs[NTOK][XPAD];     // 129 KB token tile (bf16)
  __shared__ float  Ks[NTOK][KVPAD];    // 16.6 KB
  __shared__ float  Vs[NTOK][KVPAD];    // 16.6 KB
  __shared__ float  qs[DDIM];
  __shared__ float  gs[NTOK];
  __shared__ float  Ls[NTOK];
  __shared__ float  Es[NTOK];
  __shared__ float  Aat[NTOK];
  __shared__ float  macc[NTOK];

  const int tb   = blockIdx.x;          // t*32 + b
  const int tid  = threadIdx.x;
  const int lane = tid & 31;
  const int wv   = tid >> 5;            // 0..7
  const int l15  = lane & 15;
  const int lhalf = (lane >> 4) & 1;

  // ---- stage token tile (f32 global -> bf16 LDS), q row, gate ----
  const float* tokbase = tokens + (size_t)tb * NTOK * DDIM;
  for (int e = tid; e < NTOK * 256; e += 256) {
    const int row = e >> 8;
    const int c4  = (e & 255) << 2;
    v4f f = *(const v4f*)(tokbase + row * DDIM + c4);
    __bf16* p = &Xs[row][c4];
    p[0] = (__bf16)f[0]; p[1] = (__bf16)f[1];
    p[2] = (__bf16)f[2]; p[3] = (__bf16)f[3];
  }
  for (int i = tid; i < DDIM; i += 256) qs[i] = Qout[(size_t)tb * DDIM + i];
  if (tid < NTOK) { gs[tid] = gate[tb * NTOK + tid]; macc[tid] = 0.0f; }
  __syncthreads();

  const __bf16* WT = (wv < 4) ? WkT : WvT;
  const float*  bb = (wv < 4) ? bk  : bv;
  float* KV = (wv < 4) ? &Ks[0][0] : &Vs[0][0];
  const int nt   = wv & 3;              // column tile owned by this wave
  const int ncol = nt * 16 + l15;       // per-lane output column within head
  const int rmo  = (lane >= 16) ? 8 : 0;

  for (int h = 0; h < NHEADS; ++h) {
    // ---- K_h / V_h = Xs(bf16) @ WT[:, h*64 .. h*64+63] via WMMA ----
    v8f acc[4];                          // acc[mt]: row tiles 0..3
#pragma unroll
    for (int i = 0; i < 4; ++i)
#pragma unroll
      for (int j = 0; j < 8; ++j) acc[i][j] = 0.0f;

    const __bf16* bcol = WT + (size_t)(h * 64 + ncol) * DDIM;
    for (int kb = 0; kb < DDIM; kb += 32) {
      const int kOff = kb + lhalf * 8;
      const int kpos = kb + ((lane >= 16) ? 8 : 0);
      v16bf bfrag = load_b_frag(bcol + kpos);   // unique per wave
#pragma unroll
      for (int mt = 0; mt < 4; ++mt) {
        v16bf afrag = bf_combine(*(const v8bf*)&Xs[mt * 16 + l15][kOff],
                                 *(const v8bf*)&Xs[mt * 16 + l15][kOff + 16]);
        acc[mt] = __builtin_amdgcn_wmma_f32_16x16x32_bf16(
            false, afrag, false, bfrag, (short)0, acc[mt], false, false);
      }
    }

    __syncthreads();   // previous head finished reading Ks/Vs/Aat
    {
      const float b = bb[h * 64 + ncol];
#pragma unroll
      for (int mt = 0; mt < 4; ++mt)
#pragma unroll
        for (int r = 0; r < 8; ++r)
          KV[(mt * 16 + r + rmo) * KVPAD + ncol] = acc[mt][r] + b;
    }
    __syncthreads();

    // ---- logits: q_h . K_h / sqrt(64) + gate ----
    if (tid < NTOK) {
      const int n = tid;
      const float* qh = &qs[h * HD];
      float dot = 0.0f;
#pragma unroll
      for (int d2 = 0; d2 < HD; ++d2) dot += qh[d2] * Ks[n][d2];
      const float lg = dot * 0.125f + gs[n];
      logits_out[(size_t)(tb * NHEADS + h) * NTOK + n] = lg;
      Ls[n] = lg;
    }
    __syncthreads();
    if (tid < NTOK) {
      float mx = -3.4e38f;
      for (int i = 0; i < NTOK; ++i) mx = fmaxf(mx, Ls[i]);  // LDS broadcast
      Es[tid] = __expf(Ls[tid] - mx);
    }
    __syncthreads();
    if (tid < NTOK) {
      float s = 0.0f;
      for (int i = 0; i < NTOK; ++i) s += Es[i];
      const float a = Es[tid] / s;
      Aat[tid] = a;
      macc[tid] += a;
    }
    __syncthreads();
    // ---- attended_h[d] = sum_n attn[n] * V_h[n][d] ----
    if (tid < NTOK) {
      const int d2 = tid;
      float o = 0.0f;
#pragma unroll
      for (int n2 = 0; n2 < NTOK; ++n2) o += Aat[n2] * Vs[n2][d2];
      attended[(size_t)tb * DDIM + h * HD + d2] = o;
    }
    // next iteration's pre-store barrier protects Vs/Aat reuse
  }
  __syncthreads();
  if (tid < NTOK) mean_out[tb * NTOK + tid] = macc[tid] * (1.0f / 16.0f);
}

// ---------------------------------------------------------------------------
extern "C" void kernel_launch(void* const* d_in, const int* in_sizes, int n_in,
                              void* d_out, int out_size, void* d_ws, size_t ws_size,
                              hipStream_t stream) {
  (void)in_sizes; (void)n_in; (void)out_size; (void)ws_size;
  const float* query  = (const float*)d_in[0];
  const float* tokens = (const float*)d_in[1];
  const float* gate   = (const float*)d_in[2];
  const float* Wq     = (const float*)d_in[3];
  const float* bq     = (const float*)d_in[4];
  const float* Wk     = (const float*)d_in[5];
  const float* bk     = (const float*)d_in[6];
  const float* Wv     = (const float*)d_in[7];
  const float* bv     = (const float*)d_in[8];

  char* ws = (char*)d_ws;
  __bf16* WqT  = (__bf16*)(ws);                          // 2 MB
  __bf16* WkT  = (__bf16*)(ws + (size_t)(1u << 21));     // 2 MB
  __bf16* WvT  = (__bf16*)(ws + (size_t)(2u << 21));     // 2 MB
  float*  Qout = (float*)(ws + (size_t)(3u << 21));      // 4 MB

  float* out        = (float*)d_out;
  float* attended   = out;                 // [1024,1024]
  float* mean_out   = out + 1048576;       // [1024,64]
  float* logits_out = out + 1114112;       // [1024,16,64]

  transpose_convert<<<dim3(32, 32, 3), dim3(32, 8), 0, stream>>>(
      Wq, Wk, Wv, WqT, WkT, WvT);
  qproj_kernel<<<dim3(16, 16), 128, 0, stream>>>(query, WqT, bq, Qout);
  attn_kernel<<<1024, 256, 0, stream>>>(tokens, gate, WkT, WvT, bk, bv, Qout,
                                        attended, mean_out, logits_out);
}
